// GraphPropagationBlock_2284922602204
// MI455X (gfx1250) — matrix operand order repaired and checked by reading, not compiled, not run on hardware
//
#include <hip/hip_runtime.h>
#include <hip/hip_bf16.h>
#include <math.h>

typedef __attribute__((ext_vector_type(16))) _Float16 v16h;
typedef __attribute__((ext_vector_type(8)))  _Float16 v8h;
typedef __attribute__((ext_vector_type(8)))  float    v8f;
typedef __attribute__((ext_vector_type(4)))  unsigned int u32x4;
typedef __attribute__((ext_vector_type(8)))  int i32x8;
typedef __attribute__((ext_vector_type(4)))  int i32x4;

#define BATCH 64
#define NTOK  197
#define NPAD  224
#define CDIM  768
#define HEADS 12
#define HDIM  64
#define NKEPT 147
#define NELIM 49
#define NOUT  148
#define M1    (BATCH*NTOK)   // 12608
#define M2    (BATCH*NOUT)   // 9472
#define KCH   32             // K chunk per TDM stage

// ---------------- WMMA fragment helpers (CDNA5 16x16x32 f16) ----------------
__device__ __forceinline__ v16h load_a_frag(const _Float16* __restrict__ A, int lda,
                                            int row0, int k0) {
  int lane = threadIdx.x & 31;
  int r  = row0 + (lane & 15);
  int kb = k0 + ((lane >> 4) << 4);           // +0 or +16
  const _Float16* p = A + (size_t)r * lda + kb;
  v8h lo = *(const v8h*)p;                    // K = kb..kb+7
  v8h hi = *(const v8h*)(p + 8);              // K = kb+8..kb+15
  v16h out;
#pragma unroll
  for (int i = 0; i < 8; ++i) { out[i] = lo[i]; out[i + 8] = hi[i]; }
  return out;
}

// B fragment from global [N][K] pre-transposed weights
__device__ __forceinline__ v16h load_b_frag(const _Float16* __restrict__ Wt, int ldk,
                                            int col0, int k0) {
  int lane = threadIdx.x & 31;
  int c  = col0 + (lane & 15);
  int kb = k0 + ((lane >> 4) << 4);           // +0 or +16
  return *(const v16h*)(Wt + (size_t)c * ldk + kb);
}

// B fragment from an LDS-staged [256 cols][KCH halves] tile
__device__ __forceinline__ v16h load_b_frag_lds(const _Float16* sbuf, int col0) {
  int lane = threadIdx.x & 31;
  const _Float16* p = sbuf + (size_t)(col0 + (lane & 15)) * KCH + ((lane >> 4) << 4);
  v8h lo = *(const v8h*)p;        // 16B ds_load
  v8h hi = *(const v8h*)(p + 8);  // 16B ds_load
  v16h out;
#pragma unroll
  for (int i = 0; i < 8; ++i) { out[i] = lo[i]; out[i + 8] = hi[i]; }
  return out;
}

__device__ __forceinline__ v8f wmma_f16(v16h a, v16h b, v8f c) {
  return __builtin_amdgcn_wmma_f32_16x16x32_f16(false, a, false, b, (short)0, c,
                                                false, false);
}

__device__ __forceinline__ float gelu_exact(float x) {
  return 0.5f * x * (1.0f + erff(x * 0.70710678118654752f));
}

// ---------------- Tensor Data Mover: stage a 256x32-half B tile into LDS ----------------
// D# per CDNA5 ISA ch.8: group0 {count,lds_addr,global_addr,type=2},
// group1 {data_size=2B, tensor_dim0=K, tensor_dim1=rows, tile 32x256, stride0=K}.
__device__ __forceinline__ void tdm_load_b_tile(const _Float16* gsrc,
                                                unsigned lds_addr, int K) {
  unsigned long long ga = (unsigned long long)(uintptr_t)gsrc;
  u32x4 g0;
  g0[0] = 1u;                                             // count=1, user D#
  g0[1] = lds_addr;                                       // LDS byte address
  g0[2] = (unsigned)(ga & 0xffffffffu);                   // global_addr[31:0]
  g0[3] = (unsigned)((ga >> 32) & 0x01ffffffu) | (2u << 30); // addr[56:32], type=2
  unsigned td0 = (unsigned)K;                             // tensor_dim0 (halves)
  unsigned td1 = 65536u;                                  // tensor_dim1 (rows avail)
  unsigned long long s0 = (unsigned long long)K;          // tensor_dim0_stride
  i32x8 g1;
  g1[0] = (int)(1u << 16);                                // data_size=1 (2 bytes)
  g1[1] = (int)((td0 & 0xffffu) << 16);                   // tensor_dim0 lo
  g1[2] = (int)((td0 >> 16) | ((td1 & 0xffffu) << 16));   // dim0 hi | dim1 lo
  g1[3] = (int)((td1 >> 16) | ((unsigned)KCH << 16));     // dim1 hi | tile_dim0=32
  g1[4] = (int)(256u);                                    // tile_dim1=256, tile_dim2=0
  g1[5] = (int)(s0 & 0xffffffffu);                        // stride0 lo
  g1[6] = (int)((s0 >> 32) & 0xffffu);                    // stride0 hi | stride1 lo=0
  g1[7] = 0;
  i32x4 z4 = {0, 0, 0, 0};
  i32x8 z8 = {0, 0, 0, 0, 0, 0, 0, 0};
  // clang-23 / therock-10.0 form: (g0, g1, g2, g3, g2b, cpol)
  __builtin_amdgcn_tensor_load_to_lds(g0, g1, z4, z4, z8, 0);
}

// ---------------- shared GEMM core: TDM-staged B, double-buffered LDS ----------------
__device__ __forceinline__ void gemm_core(const _Float16* __restrict__ A,
                                          const _Float16* __restrict__ Wt,
                                          _Float16* sB, int K,
                                          int rowTile, int colBlock, v8f acc[4]) {
  int wave = threadIdx.x >> 5;
  int colWave = (wave & 3) * 64;
  int nc = K / KCH;
  bool issuer = (threadIdx.x < 32);       // wave 0 drives the TDM
  if (issuer)
    tdm_load_b_tile(Wt + (size_t)colBlock * K, (unsigned)(uintptr_t)sB, K);
  for (int c = 0; c < nc; ++c) {
    int k0 = c * KCH;
    if (c + 1 < nc) {
      if (issuer)
        tdm_load_b_tile(Wt + (size_t)colBlock * K + (k0 + KCH),
                        (unsigned)(uintptr_t)(sB + ((c + 1) & 1) * (256 * KCH)), K);
      __builtin_amdgcn_s_wait_tensorcnt(1);  // previous stage complete
    } else {
      __builtin_amdgcn_s_wait_tensorcnt(0);
    }
    __syncthreads();                         // publish LDS tile to all waves
    const _Float16* sbuf = sB + (c & 1) * (256 * KCH);
    __builtin_prefetch((const void*)(A + (size_t)rowTile * K + k0 + 64), 0, 1);
    v16h a = load_a_frag(A, K, rowTile, k0);
#pragma unroll
    for (int t = 0; t < 4; ++t) {
      v16h b = load_b_frag_lds(sbuf, colWave + t * 16);
      acc[t] = wmma_f16(a, b, acc[t]);
    }
    __syncthreads();                         // buffer reuse fence
  }
}

// ---------------- utility kernels ----------------
__global__ void zero_h_kernel(_Float16* p, size_t n) {
  size_t i = (size_t)blockIdx.x * 256 + threadIdx.x;
  if (i < n) p[i] = (_Float16)0.0f;
}
__global__ void zero_f_kernel(float* p, size_t n) {
  size_t i = (size_t)blockIdx.x * 256 + threadIdx.x;
  if (i < n) p[i] = 0.0f;
}

// f32 [K][N] -> f16 [N][K]
__global__ void wtrans_kernel(const float* __restrict__ W, _Float16* __restrict__ Wt,
                              int K, int N) {
  size_t i = (size_t)blockIdx.x * 256 + threadIdx.x;
  if (i >= (size_t)N * K) return;
  int n = (int)(i / K), k = (int)(i % K);
  Wt[i] = (_Float16)W[(size_t)k * N + n];
}

// LayerNorm over C=768, f32 in -> f16 out
__global__ __launch_bounds__(256) void ln_kernel(const float* __restrict__ x,
                                                 const float* __restrict__ g,
                                                 const float* __restrict__ bta,
                                                 _Float16* __restrict__ out) {
  const int C = CDIM;
  __shared__ float red[256];
  size_t row = blockIdx.x;
  const float* xr = x + row * C;
  float s = 0.f;
  for (int c = threadIdx.x; c < C; c += 256) s += xr[c];
  red[threadIdx.x] = s; __syncthreads();
  for (int o = 128; o > 0; o >>= 1) {
    if (threadIdx.x < o) red[threadIdx.x] += red[threadIdx.x + o];
    __syncthreads();
  }
  float mean = red[0] * (1.0f / C);
  __syncthreads();
  float s2 = 0.f;
  for (int c = threadIdx.x; c < C; c += 256) { float d = xr[c] - mean; s2 += d * d; }
  red[threadIdx.x] = s2; __syncthreads();
  for (int o = 128; o > 0; o >>= 1) {
    if (threadIdx.x < o) red[threadIdx.x] += red[threadIdx.x + o];
    __syncthreads();
  }
  float rstd = rsqrtf(red[0] * (1.0f / C) + 1e-5f);
  for (int c = threadIdx.x; c < C; c += 256)
    out[row * C + c] = (_Float16)((xr[c] - mean) * rstd * g[c] + bta[c]);
}

// ---------------- generic WMMA GEMM: out = [act](A @ Wt^T + bias) [+ resid] ----------------
constexpr int GM_RES = 1, GM_GELU = 2, GM_OUTF16 = 4;

template <int MODE>
__global__ __launch_bounds__(256) void gemm_f16(const _Float16* __restrict__ A,
                                                const _Float16* __restrict__ Wt,
                                                const float* __restrict__ bias,
                                                const float* __restrict__ resid,
                                                float* __restrict__ outF,
                                                _Float16* __restrict__ outH,
                                                int K, int N) {
  __shared__ _Float16 sB[2 * 256 * KCH];     // 32 KB double-buffered B tile
  int wave = threadIdx.x >> 5;
  int rowTile = blockIdx.x * 32 + (wave >> 2) * 16;
  int colBlock = blockIdx.y * 256;
  int colTile = colBlock + (wave & 3) * 64;
  v8f acc[4] = {};
  gemm_core(A, Wt, sB, K, rowTile, colBlock, acc);
  int lane = threadIdx.x & 31;
  int n0 = lane & 15;
  int rb = rowTile + ((lane >> 4) << 3);
#pragma unroll
  for (int t = 0; t < 4; ++t) {
    int ncol = colTile + t * 16 + n0;
    float bv = bias[ncol];
#pragma unroll
    for (int r = 0; r < 8; ++r) {
      size_t row = (size_t)(rb + r);
      float v = acc[t][r] + bv;
      if (MODE & GM_GELU) v = gelu_exact(v);
      if (MODE & GM_RES)  v += resid[row * N + ncol];
      if (MODE & GM_OUTF16) outH[row * N + ncol] = (_Float16)v;
      else                  outF[row * N + ncol] = v;
    }
  }
}

// ---------------- QKV GEMM with scatter epilogue ----------------
// q: [B][H][224][64] (scaled by 1/8), k: [B][H][224][64], vt: [B][H][64][224]
__global__ __launch_bounds__(256) void gemm_qkv(const _Float16* __restrict__ A,
                                                const _Float16* __restrict__ Wt,
                                                const float* __restrict__ bias,
                                                _Float16* __restrict__ q,
                                                _Float16* __restrict__ k,
                                                _Float16* __restrict__ vt) {
  __shared__ _Float16 sB[2 * 256 * KCH];
  const int K = CDIM;
  int wave = threadIdx.x >> 5;
  int rowTile = blockIdx.x * 32 + (wave >> 2) * 16;
  int colBlock = blockIdx.y * 256;
  int colTile = colBlock + (wave & 3) * 64;
  v8f acc[4] = {};
  gemm_core(A, Wt, sB, K, rowTile, colBlock, acc);
  int lane = threadIdx.x & 31;
  int n0 = lane & 15;
  int rbase = rowTile + ((lane >> 4) << 3);
#pragma unroll
  for (int t = 0; t < 4; ++t) {
    int col = colTile + t * 16 + n0;
    int which = col / CDIM, j = col % CDIM;
    int hh = j >> 6, d = j & 63;
    float bv = bias[col];
#pragma unroll
    for (int r = 0; r < 8; ++r) {
      int row = rbase + r;
      int b = row / NTOK, tok = row % NTOK;
      float v = acc[t][r] + bv;
      size_t bh = (size_t)(b * HEADS + hh);
      if (which == 0)      q[(bh * NPAD + tok) * HDIM + d] = (_Float16)(v * 0.125f);
      else if (which == 1) k[(bh * NPAD + tok) * HDIM + d] = (_Float16)v;
      else                 vt[(bh * HDIM + d) * NPAD + tok] = (_Float16)v;
    }
  }
}

// ---------------- attention: softmax(QK^T + log ts) V, plus token scores ----------------
__global__ __launch_bounds__(128) void attn_kernel(const _Float16* __restrict__ qbuf,
                                                   const _Float16* __restrict__ kbuf,
                                                   const _Float16* __restrict__ vtbuf,
                                                   const float* __restrict__ ts,
                                                   float* __restrict__ tr1,
                                                   float* __restrict__ tr2,
                                                   _Float16* __restrict__ attn_out) {
  __shared__ float slog[4][16][NPAD];   // per-wave logits/probs
  __shared__ float sinv[4][16];         // per-row 1/l
  int wave = threadIdx.x >> 5, lane = threadIdx.x & 31;
  int bh = blockIdx.x;
  int b = bh / HEADS, h = bh % HEADS;
  int qb = blockIdx.y * 4 + wave;
  bool active = (qb < 13);              // 13 * 16 = 208 >= 197

  const _Float16* Q  = qbuf  + ((size_t)bh * NPAD + qb * 16) * HDIM;
  const _Float16* Kb = kbuf  + (size_t)bh * NPAD * HDIM;
  const _Float16* Vt = vtbuf + (size_t)bh * HDIM * NPAD;

  int n0 = lane & 15;
  int rb = (lane >> 4) << 3;
  int row = lane & 15;
  int half = lane >> 4;

  if (active) {
    v16h a0 = load_a_frag(Q, HDIM, 0, 0);
    v16h a1 = load_a_frag(Q, HDIM, 0, 32);
    for (int kt = 0; kt < 7; ++kt) {
#pragma unroll
      for (int sub = 0; sub < 2; ++sub) {
        int col0 = kt * 32 + sub * 16;
        v8f s = {};
        s = wmma_f16(a0, load_b_frag(Kb, HDIM, col0, 0), s);
        s = wmma_f16(a1, load_b_frag(Kb, HDIM, col0, 32), s);
        int ncol = col0 + n0;
        float biasv = (ncol < NTOK) ? __logf(ts[b * NTOK + ncol]) : 0.0f;
#pragma unroll
        for (int r = 0; r < 8; ++r) {
          float v = (ncol < NTOK) ? (s[r] + biasv) : -1e30f;
          slog[wave][rb + r][ncol] = v;
        }
      }
    }
  }
  __syncthreads();

  float invl = 0.f;
  if (active) {
    float m = -1e30f;
    for (int i = 0; i < 112; ++i) m = fmaxf(m, slog[wave][row][half * 112 + i]);
    m = fmaxf(m, __shfl_xor(m, 16, 32));
    float l = 0.f;
    for (int i = 0; i < 112; ++i) {
      float p = __expf(slog[wave][row][half * 112 + i] - m);
      slog[wave][row][half * 112 + i] = p;
      l += p;
    }
    l += __shfl_xor(l, 16, 32);
    invl = 1.0f / l;
    sinv[wave][row] = invl;
  }
  __syncthreads();

  if (active) {
    int qrow = qb * 16 + row;
    // tr1: diagonal (tokens 1..196), mean over heads
    if (qrow >= 1 && qrow < NTOK) {
      int col = qrow;
      if (col >= half * 112 && col < half * 112 + 112)
        atomicAdd(&tr1[b * 196 + (qrow - 1)],
                  slog[wave][row][col] * invl * (1.0f / HEADS));
    }
    // tr2: column sums over all queries (keys 1..196), mean over heads
    if (qrow < NTOK) {
      for (int i = 0; i < 112; ++i) {
        int col = half * 112 + i;
        if (col >= 1 && col < NTOK)
          atomicAdd(&tr2[b * 196 + (col - 1)],
                    slog[wave][row][col] * invl * (1.0f / HEADS));
      }
    }

    // P @ V^T  (P unnormalized; fold 1/l at store)
    v8f o[4] = {};
    for (int kt = 0; kt < 7; ++kt) {
      v16h pa;
      int kb2 = (lane >> 4) << 4;
#pragma unroll
      for (int e = 0; e < 8; ++e) {
        pa[e]     = (_Float16)slog[wave][row][kt * 32 + kb2 + e];
        pa[e + 8] = (_Float16)slog[wave][row][kt * 32 + kb2 + 8 + e];
      }
#pragma unroll
      for (int t = 0; t < 4; ++t)
        o[t] = wmma_f16(pa, load_b_frag(Vt, NPAD, t * 16, kt * 32), o[t]);
    }
#pragma unroll
    for (int t = 0; t < 4; ++t) {
      int dcol = t * 16 + n0;
#pragma unroll
      for (int r = 0; r < 8; ++r) {
        int qrow2 = qb * 16 + rb + r;
        if (qrow2 < NTOK)
          attn_out[((size_t)b * NTOK + qrow2) * CDIM + h * HDIM + dcol] =
              (_Float16)(o[t][r] * sinv[wave][rb + r]);
      }
    }
  }
}

// ---------------- per-batch bitonic argsort of tr1*tr2 (descending) ----------------
__global__ __launch_bounds__(256) void sort_kernel(const float* __restrict__ tr1,
                                                   const float* __restrict__ tr2,
                                                   int* __restrict__ kept,
                                                   int* __restrict__ elim) {
  __shared__ float v[256];
  __shared__ int id[256];
  int b = blockIdx.x, t = threadIdx.x;
  v[t] = (t < 196) ? tr1[b * 196 + t] * tr2[b * 196 + t] : -1e30f;
  id[t] = t;
  __syncthreads();
  for (int kk = 2; kk <= 256; kk <<= 1) {
    for (int j = kk >> 1; j > 0; j >>= 1) {
      int ixj = t ^ j;
      if (ixj > t) {
        bool desc = ((t & kk) == 0);
        float va = v[t], vb = v[ixj];
        bool sw = desc ? (va < vb) : (va > vb);
        if (sw) {
          int ia = id[t];
          v[t] = vb; v[ixj] = va;
          id[t] = id[ixj]; id[ixj] = ia;
        }
      }
      __syncthreads();
    }
  }
  if (t < NKEPT) kept[b * NKEPT + t] = id[t];
  else if (t < 196) elim[b * NELIM + (t - NKEPT)] = id[t];
}

// ---------------- gather + scale eliminated tokens ----------------
__global__ __launch_bounds__(256) void gather_elim(const float* __restrict__ x1,
                                                   const float* __restrict__ ts,
                                                   const int* __restrict__ elim,
                                                   float* __restrict__ xes,
                                                   float* __restrict__ tselim) {
  int b = blockIdx.x, e = blockIdx.y;
  int ie = elim[b * NELIM + e];
  float tse = ts[b * NTOK + 1 + ie];
  if (threadIdx.x == 0) tselim[b * NELIM + e] = tse;
  const float* src = x1 + ((size_t)b * NTOK + 1 + ie) * CDIM;
  float* dst = xes + ((size_t)b * NELIM + e) * CDIM;
  for (int c = threadIdx.x; c < CDIM; c += 256) dst[c] = src[c] * tse;
}

// ---------------- graph propagation ----------------
__global__ __launch_bounds__(256) void propagate(const float* __restrict__ x1,
                                                 const float* __restrict__ graph,
                                                 const float* __restrict__ ts,
                                                 const int* __restrict__ kept,
                                                 const int* __restrict__ elim,
                                                 const float* __restrict__ xes,
                                                 const float* __restrict__ tselim,
                                                 float* __restrict__ x2,
                                                 float* __restrict__ out_wkept,
                                                 float* __restrict__ out_ts) {
  __shared__ float wp[NELIM];
  __shared__ float tsnew_s;
  int b = blockIdx.x, i = blockIdx.y;
  int ik = kept[b * NKEPT + i];
  const float* grow = graph + ((size_t)b * 196 + ik) * 196;
  if (threadIdx.x < NELIM) wp[threadIdx.x] = grow[elim[b * NELIM + threadIdx.x]];
  __syncthreads();
  for (int j = threadIdx.x; j < NKEPT; j += 256)
    out_wkept[((size_t)b * NKEPT + i) * NKEPT + j] = grow[kept[b * NKEPT + j]];
  float tsk = ts[b * NTOK + 1 + ik];
  if (threadIdx.x == 0) {
    float dot = 0.f;
    for (int e = 0; e < NELIM; ++e) dot += wp[e] * tselim[b * NELIM + e];
    float tn = tsk + dot;
    tsnew_s = tn;
    out_ts[b * NOUT + 1 + i] = tn;
    if (i == 0) out_ts[b * NOUT] = ts[b * NTOK];  // cls scale
  }
  __syncthreads();
  float tsnew = tsnew_s;
  const float* xk = x1 + ((size_t)b * NTOK + 1 + ik) * CDIM;
  const float* xe = xes + (size_t)b * NELIM * CDIM;
  float* dst = x2 + ((size_t)b * NOUT + 1 + i) * CDIM;
  for (int c = threadIdx.x; c < CDIM; c += 256) {
    float acc = xk[c] * tsk;
    for (int e = 0; e < NELIM; ++e) acc += wp[e] * xe[(size_t)e * CDIM + c];
    dst[c] = acc / tsnew;
  }
  if (i == 0) {  // cls row passthrough
    const float* src = x1 + (size_t)b * NTOK * CDIM;
    float* d0 = x2 + (size_t)b * NOUT * CDIM;
    for (int c = threadIdx.x; c < CDIM; c += 256) d0[c] = src[c];
  }
}

// ---------------- host launch ----------------
extern "C" void kernel_launch(void* const* d_in, const int* in_sizes, int n_in,
                              void* d_out, int out_size, void* d_ws, size_t ws_size,
                              hipStream_t stream) {
  const float* x      = (const float*)d_in[0];
  const float* graph  = (const float*)d_in[1];
  const float* ts     = (const float*)d_in[2];
  const float* ln1_g  = (const float*)d_in[3];
  const float* ln1_b  = (const float*)d_in[4];
  const float* w_qkv  = (const float*)d_in[5];
  const float* b_qkv  = (const float*)d_in[6];
  const float* w_proj = (const float*)d_in[7];
  const float* b_proj = (const float*)d_in[8];
  const float* ln2_g  = (const float*)d_in[9];
  const float* ln2_b  = (const float*)d_in[10];
  const float* w_fc1  = (const float*)d_in[11];
  const float* b_fc1  = (const float*)d_in[12];
  const float* w_fc2  = (const float*)d_in[13];
  const float* b_fc2  = (const float*)d_in[14];

  float* outX  = (float*)d_out;                                      // [64][148][768]
  float* outW  = outX + (size_t)BATCH * NOUT * CDIM;                 // [64][147][147]
  float* outTS = outW + (size_t)BATCH * NKEPT * NKEPT;               // [64][148]

  char* p = (char*)d_ws;
  auto alloc = [&](size_t bytes) -> void* {
    void* r = (void*)p;
    p += (bytes + 255) & ~(size_t)255;
    return r;
  };
  _Float16* hA     = (_Float16*)alloc((size_t)M1 * CDIM * 2);
  _Float16* wqkvT  = (_Float16*)alloc((size_t)3 * CDIM * CDIM * 2);
  _Float16* wprojT = (_Float16*)alloc((size_t)CDIM * CDIM * 2);
  _Float16* wfc1T  = (_Float16*)alloc((size_t)4 * CDIM * CDIM * 2);
  _Float16* wfc2T  = (_Float16*)alloc((size_t)4 * CDIM * CDIM * 2);
  size_t qkvHalves = (size_t)BATCH * HEADS * NPAD * HDIM;
  _Float16* qbuf   = (_Float16*)alloc(qkvHalves * 3 * 2);
  _Float16* kbuf   = qbuf + qkvHalves;
  _Float16* vtbuf  = kbuf + qkvHalves;
  _Float16* attn_o = (_Float16*)alloc((size_t)M1 * CDIM * 2);
  float* x1        = (float*)alloc((size_t)M1 * CDIM * 4);
  float* tr1       = (float*)alloc((size_t)BATCH * 196 * 2 * 4);
  float* tr2       = tr1 + (size_t)BATCH * 196;
  int*   kept      = (int*)alloc((size_t)BATCH * NKEPT * 4);
  int*   elim      = (int*)alloc((size_t)BATCH * NELIM * 4);
  float* xes       = (float*)alloc((size_t)BATCH * NELIM * CDIM * 4);
  float* tselim    = (float*)alloc((size_t)BATCH * NELIM * 4);
  float* x2        = (float*)alloc((size_t)M2 * CDIM * 4);
  _Float16* h2     = (_Float16*)alloc((size_t)M2 * CDIM * 2);
  _Float16* act    = (_Float16*)alloc((size_t)M2 * 4 * CDIM * 2);

  {
    size_t n;
    n = (size_t)CDIM * 3 * CDIM;
    wtrans_kernel<<<(n + 255) / 256, 256, 0, stream>>>(w_qkv, wqkvT, CDIM, 3 * CDIM);
    n = (size_t)CDIM * CDIM;
    wtrans_kernel<<<(n + 255) / 256, 256, 0, stream>>>(w_proj, wprojT, CDIM, CDIM);
    n = (size_t)CDIM * 4 * CDIM;
    wtrans_kernel<<<(n + 255) / 256, 256, 0, stream>>>(w_fc1, wfc1T, CDIM, 4 * CDIM);
    wtrans_kernel<<<(n + 255) / 256, 256, 0, stream>>>(w_fc2, wfc2T, 4 * CDIM, CDIM);
  }
  {
    size_t n = qkvHalves * 3;
    zero_h_kernel<<<(n + 255) / 256, 256, 0, stream>>>(qbuf, n);
    size_t m = (size_t)BATCH * 196 * 2;
    zero_f_kernel<<<(m + 255) / 256, 256, 0, stream>>>(tr1, m);
  }
  ln_kernel<<<M1, 256, 0, stream>>>(x, ln1_g, ln1_b, hA);
  gemm_qkv<<<dim3(M1 / 32, (3 * CDIM) / 256), 256, 0, stream>>>(hA, wqkvT, b_qkv,
                                                                qbuf, kbuf, vtbuf);
  attn_kernel<<<dim3(BATCH * HEADS, 4), 128, 0, stream>>>(qbuf, kbuf, vtbuf, ts,
                                                          tr1, tr2, attn_o);
  gemm_f16<GM_RES><<<dim3(M1 / 32, CDIM / 256), 256, 0, stream>>>(
      attn_o, wprojT, b_proj, x, x1, nullptr, CDIM, CDIM);
  sort_kernel<<<BATCH, 256, 0, stream>>>(tr1, tr2, kept, elim);
  gather_elim<<<dim3(BATCH, NELIM), 256, 0, stream>>>(x1, ts, elim, xes, tselim);
  propagate<<<dim3(BATCH, NKEPT), 256, 0, stream>>>(x1, graph, ts, kept, elim, xes,
                                                    tselim, x2, outW, outTS);
  ln_kernel<<<M2, 256, 0, stream>>>(x2, ln2_g, ln2_b, h2);
  gemm_f16<GM_GELU | GM_OUTF16><<<dim3(M2 / 32, (4 * CDIM) / 256), 256, 0, stream>>>(
      h2, wfc1T, b_fc1, nullptr, nullptr, act, CDIM, 4 * CDIM);
  gemm_f16<GM_RES><<<dim3(M2 / 32, CDIM / 256), 256, 0, stream>>>(
      act, wfc2T, b_fc2, x2, outX, nullptr, 4 * CDIM, CDIM);
}
